// QuantizedLoRALinear_85890755985991
// MI455X (gfx1250) — compile-verified, matrix operand order checked by reading, not run on hardware
//
#include <hip/hip_runtime.h>
#include <stdint.h>

// ---------------------------------------------------------------------------
// QLoRA linear for MI455X (gfx1250): NF4-dequant -> bf16 WMMA GEMM + fused
// bias + rank-8 LoRA epilogue.  Workspace layout (96.25 MB):
//   [0)                 wbf : 4096*4096 bf16  (32 MB)  dequantized weight
//   [32MB)              xbf : M*4096    bf16  (64 MB)  x cast to bf16
//   [96MB)           hidden : M*8       f32  (256 KB)  x @ lora_A^T
// GEMM global->LDS staging uses GLOBAL_LOAD_ASYNC_TO_LDS_B128 (ASYNCcnt),
// eliminating VGPR staging (and the scratch spills it caused in round 1).
// ---------------------------------------------------------------------------

#define IN_F   4096
#define OUT_F  4096
#define RANK   8
#define SCALING 2.0f

typedef __attribute__((ext_vector_type(8)))  __bf16 v8bf;
typedef __attribute__((ext_vector_type(16))) __bf16 v16bf;
typedef __attribute__((ext_vector_type(8)))  float  v8f;

__constant__ float NF4_LUT[16] = {
    -1.0f, -0.6961928009986877f, -0.5250730514526367f, -0.39491748809814453f,
    -0.28444138169288635f, -0.18477343022823334f, -0.09105003625154495f, 0.0f,
    0.07958029955625534f, 0.16093020141124725f, 0.24611230194568634f,
    0.33791524171829224f, 0.44070982933044434f, 0.5626170039176941f,
    0.7229568362236023f, 1.0f};

__device__ __forceinline__ unsigned short f32_to_bf16(float f) {
    union { float f; unsigned u; } v; v.f = f;
    unsigned r = v.u + 0x7FFFu + ((v.u >> 16) & 1u);  // round-nearest-even
    return (unsigned short)(r >> 16);
}

__device__ __forceinline__ void wait_asynccnt0() {
#if __has_builtin(__builtin_amdgcn_s_wait_asynccnt)
    __builtin_amdgcn_s_wait_asynccnt(0);
#else
    asm volatile("s_wait_asynccnt 0x0" ::: "memory");
#endif
}

// async global(16B) -> LDS copy; lds_addr = low 32 bits of generic pointer
// (ISA 10.2).  Tracked by ASYNCcnt.
__device__ __forceinline__ void async_copy16(const void* gptr, void* lptr) {
    unsigned lds = (unsigned)(size_t)lptr;
    asm volatile("global_load_async_to_lds_b128 %0, %1, off"
                 :: "v"(lds), "v"(gptr) : "memory");
}

// --------------------------- 1) NF4 dequant --------------------------------
__global__ __launch_bounds__(256)
void nf4_dequant_bf16(const int* __restrict__ codes,
                      const float* __restrict__ scales,
                      unsigned short* __restrict__ wbf) {
    const int t = blockIdx.x * blockDim.x + threadIdx.x;   // 8 elems/thread
    const float s = scales[t >> 3];                        // (t*8)>>6
    const int4 c0 = ((const int4*)codes)[t * 2 + 0];
    const int4 c1 = ((const int4*)codes)[t * 2 + 1];
    union { unsigned short u16[8]; uint4 v; } o;
    o.u16[0] = f32_to_bf16(NF4_LUT[c0.x & 15] * s);
    o.u16[1] = f32_to_bf16(NF4_LUT[c0.y & 15] * s);
    o.u16[2] = f32_to_bf16(NF4_LUT[c0.z & 15] * s);
    o.u16[3] = f32_to_bf16(NF4_LUT[c0.w & 15] * s);
    o.u16[4] = f32_to_bf16(NF4_LUT[c1.x & 15] * s);
    o.u16[5] = f32_to_bf16(NF4_LUT[c1.y & 15] * s);
    o.u16[6] = f32_to_bf16(NF4_LUT[c1.z & 15] * s);
    o.u16[7] = f32_to_bf16(NF4_LUT[c1.w & 15] * s);
    ((uint4*)wbf)[t] = o.v;
}

// --------------------------- 2) x -> bf16 ----------------------------------
__global__ __launch_bounds__(256)
void xcast_bf16(const float* __restrict__ x, unsigned short* __restrict__ xbf) {
    const int t = blockIdx.x * blockDim.x + threadIdx.x;   // 8 elems/thread
    const float4 a = ((const float4*)x)[t * 2 + 0];
    const float4 b = ((const float4*)x)[t * 2 + 1];
    union { unsigned short u16[8]; uint4 v; } o;
    o.u16[0] = f32_to_bf16(a.x); o.u16[1] = f32_to_bf16(a.y);
    o.u16[2] = f32_to_bf16(a.z); o.u16[3] = f32_to_bf16(a.w);
    o.u16[4] = f32_to_bf16(b.x); o.u16[5] = f32_to_bf16(b.y);
    o.u16[6] = f32_to_bf16(b.z); o.u16[7] = f32_to_bf16(b.w);
    ((uint4*)xbf)[t] = o.v;
}

// --------------------------- 3) LoRA hidden --------------------------------
__global__ __launch_bounds__(256)
void lora_hidden_kernel(const float* __restrict__ x,
                        const float* __restrict__ loraA,
                        float* __restrict__ hidden) {
    const int m   = blockIdx.x;
    const int tid = threadIdx.x;
    const float* xr = x + (size_t)m * IN_F;
    float acc[RANK];
#pragma unroll
    for (int r = 0; r < RANK; ++r) acc[r] = 0.f;
    for (int i = tid; i < IN_F; i += 256) {
        const float xv = xr[i];
#pragma unroll
        for (int r = 0; r < RANK; ++r) acc[r] += xv * loraA[r * IN_F + i];
    }
    __shared__ float red[256 * RANK];
#pragma unroll
    for (int r = 0; r < RANK; ++r) red[tid * RANK + r] = acc[r];
    __syncthreads();
    for (int s = 128; s > 0; s >>= 1) {
        if (tid < s) {
#pragma unroll
            for (int r = 0; r < RANK; ++r)
                red[tid * RANK + r] += red[(tid + s) * RANK + r];
        }
        __syncthreads();
    }
    if (tid < RANK) hidden[m * RANK + tid] = red[tid];
}

// --------------------------- 4) WMMA GEMM ----------------------------------
// C[m][n] = sum_k xbf[m][k]*wbf[n][k]  + bias[n] + 2*dot(hidden[m], loraB[n])
// Block: 128(M) x 128(N), 256 threads = 8 wave32; wave = 2x4 grid of
// v_wmma_f32_16x16x32_bf16 accumulators.  K: 64-wide tiles double-buffered
// in 64 KB LDS, filled by global_load_async_to_lds_b128.
#define MT 128
#define NT 128
#define KT 64
#define LDSK 64                    // LDS row stride in bf16 elems
#define KTILES (IN_F / KT)         // 64

__global__ __launch_bounds__(256)
void qlora_gemm_wmma(const unsigned short* __restrict__ xbf,
                     const unsigned short* __restrict__ wbf,
                     const float* __restrict__ hidden,
                     const float* __restrict__ bias,
                     const float* __restrict__ loraB,
                     float* __restrict__ out) {
    __shared__ uint4 smem[2][2][1024];   // [buf][A/B][16B-chunk]  = 64 KB

    const int tid  = threadIdx.x;
    const int lane = tid & 31;
    const int wave = tid >> 5;
    const int wr   = wave & 3;           // M sub-tile
    const int wc   = wave >> 2;          // N sub-tile
    const int mBlock = blockIdx.y * MT;
    const int nBlock = blockIdx.x * NT;

    const uint4* gA = (const uint4*)xbf; // row stride IN_F/8 = 512 uint4
    const uint4* gB = (const uint4*)wbf;
    const int row = tid >> 3;            // copy row (0..31 base)
    const int col = tid & 7;             // 16B chunk within 128B row

    v8f acc[2][4];
#pragma unroll
    for (int mf = 0; mf < 2; ++mf)
#pragma unroll
        for (int nf = 0; nf < 4; ++nf)
#pragma unroll
            for (int e = 0; e < 8; ++e) acc[mf][nf][e] = 0.f;

    // Issue async tile copy: 4 A-chunks + 4 B-chunks per thread, no VGPR data.
    auto issue_copy = [&](int buf, int kt) {
        const int k4 = kt * (KT / 8);
#pragma unroll
        for (int j = 0; j < 4; ++j) {
            const int r     = row + j * 32;
            const int chunk = tid + j * 256;
            async_copy16(&gA[(size_t)(mBlock + r) * (IN_F / 8) + k4 + col],
                         &smem[buf][0][chunk]);
            async_copy16(&gB[(size_t)(nBlock + r) * (IN_F / 8) + k4 + col],
                         &smem[buf][1][chunk]);
        }
    };
    // 16-bit A-matrix fragment per ISA 7.12.2: lane<16 -> K[0..7],K[16..23],
    // lane>=16 -> K[8..15],K[24..31]; row = lane&15.  Same pattern reused for
    // B since both source row-major [row][k] tiles (C = X * W^T).
    auto ld_frag = [&](const __bf16* tile, int row0, int kk) -> v16bf {
        const __bf16* p = tile + (size_t)(row0 + (lane & 15)) * LDSK + kk +
                          ((lane >> 4) << 3);
        v8bf lo = *(const v8bf*)(p);
        v8bf hi = *(const v8bf*)(p + 16);
        return __builtin_shufflevector(lo, hi, 0, 1, 2, 3, 4, 5, 6, 7,
                                       8, 9, 10, 11, 12, 13, 14, 15);
    };

    issue_copy(0, 0);
    wait_asynccnt0();
    __syncthreads();

    for (int kt = 0; kt < KTILES; ++kt) {
        const int buf = kt & 1;
        if (kt + 1 < KTILES) issue_copy(buf ^ 1, kt + 1);
        if (kt + 2 < KTILES) {   // global_prefetch_b8 the K+2 tile
            const int k4 = (kt + 2) * (KT / 8);
            __builtin_prefetch(&gA[(size_t)(mBlock + row) * (IN_F / 8) + k4 + col], 0, 0);
            __builtin_prefetch(&gB[(size_t)(nBlock + row) * (IN_F / 8) + k4 + col], 0, 0);
        }
        const __bf16* At = (const __bf16*)smem[buf][0];
        const __bf16* Bt = (const __bf16*)smem[buf][1];
#pragma unroll
        for (int kk = 0; kk < KT; kk += 32) {
            v16bf af[2], bfr[4];
#pragma unroll
            for (int mf = 0; mf < 2; ++mf)
                af[mf] = ld_frag(At, wr * 32 + mf * 16, kk);
#pragma unroll
            for (int nf = 0; nf < 4; ++nf)
                bfr[nf] = ld_frag(Bt, wc * 64 + nf * 16, kk);
#pragma unroll
            for (int mf = 0; mf < 2; ++mf)
#pragma unroll
                for (int nf = 0; nf < 4; ++nf)
                    acc[mf][nf] = __builtin_amdgcn_wmma_f32_16x16x32_bf16(
                        false, af[mf], false, bfr[nf],
                        (short)0, acc[mf][nf], false, false);
        }
        // each wave waits for ITS async fills of buf^1, then one barrier
        // makes the whole next tile visible to everyone.
        wait_asynccnt0();
        __syncthreads();
    }

    // Epilogue: C/D 16x16 f32 layout -> lane<16: N=lane, M=v; lane>=16:
    // N=lane-16, M=v+8.  Fuse bias + 2*hidden.loraB.
    const int mBase = mBlock + wr * 32;
    const int nBase = nBlock + wc * 64;
    const int ncol  = lane & 15;
    const int rhi   = (lane >> 4) * 8;

    float bsv[4], lb[4][RANK];
#pragma unroll
    for (int nf = 0; nf < 4; ++nf) {
        const int n = nBase + nf * 16 + ncol;
        bsv[nf] = bias[n];
#pragma unroll
        for (int r = 0; r < RANK; ++r) lb[nf][r] = loraB[n * RANK + r];
    }
#pragma unroll
    for (int mf = 0; mf < 2; ++mf) {
#pragma unroll
        for (int v = 0; v < 8; ++v) {
            const int m = mBase + mf * 16 + rhi + v;
            const float* h = hidden + (size_t)m * RANK;
            float hr[RANK];
#pragma unroll
            for (int r = 0; r < RANK; ++r) hr[r] = h[r];
#pragma unroll
            for (int nf = 0; nf < 4; ++nf) {
                float ad = 0.f;
#pragma unroll
                for (int r = 0; r < RANK; ++r) ad += hr[r] * lb[nf][r];
                out[(size_t)m * OUT_F + nBase + nf * 16 + ncol] =
                    acc[mf][nf][v] + bsv[nf] + SCALING * ad;
            }
        }
    }
}

// --------------------------- host launcher ---------------------------------
extern "C" void kernel_launch(void* const* d_in, const int* in_sizes, int n_in,
                              void* d_out, int out_size, void* d_ws, size_t ws_size,
                              hipStream_t stream) {
    const float* x       = (const float*)d_in[0];
    const int*   codes   = (const int*)d_in[1];
    const float* scales  = (const float*)d_in[2];
    const float* bias    = (const float*)d_in[3];
    const float* loraA   = (const float*)d_in[4];
    const float* loraB   = (const float*)d_in[5];
    float*       out     = (float*)d_out;

    const int M = in_sizes[0] / IN_F;                  // 8192 tokens

    unsigned short* wbf    = (unsigned short*)d_ws;                 // 32 MB
    unsigned short* xbf    = wbf + (size_t)OUT_F * IN_F;            // 64 MB
    float*          hidden = (float*)(xbf + (size_t)M * IN_F);      // 256 KB

    // 1) dequantize NF4 weight -> bf16 (L2-resident: 32 MB << 192 MB)
    nf4_dequant_bf16<<<(OUT_F * IN_F / 8) / 256, 256, 0, stream>>>(codes, scales, wbf);
    // 2) cast activations -> bf16
    xcast_bf16<<<((size_t)M * IN_F / 8) / 256, 256, 0, stream>>>(x, xbf);
    // 3) rank-8 LoRA hidden in f32
    lora_hidden_kernel<<<M, 256, 0, stream>>>(x, loraA, hidden);
    // 4) WMMA GEMM with fused bias + LoRA epilogue
    dim3 grid(OUT_F / NT, M / MT);   // 32 x 64 = 2048 blocks
    qlora_gemm_wmma<<<grid, 256, 0, stream>>>(xbf, wbf, hidden, bias, loraB, out);
}
